// SupConLoss_36352603193992
// MI455X (gfx1250) — compile-verified, matrix operand order checked by reading, not run on hardware
//
#include <hip/hip_runtime.h>
#include <hip/hip_bf16.h>

typedef _Float16 v16h __attribute__((ext_vector_type(16)));
typedef float    v8f  __attribute__((ext_vector_type(8)));
typedef int      v8i  __attribute__((ext_vector_type(8)));

#define NN   4096
#define DD   128
#define AA   512
#define TAUI (1.0f/0.07f)
#define THR  0.3f

union HU { v16h v; unsigned u[8]; };
union IU { v8i  v; unsigned u[8]; };

__device__ __forceinline__ int baseF(int v, int h) {
    return ((v >> 2) << 4) + ((v & 3) << 1) + (h << 3);
}
__device__ __forceinline__ int baseI(int v, int h) {
    return ((v >> 1) << 4) + ((v & 1) << 2) + (h << 3);
}

__global__ __launch_bounds__(128) void knorm(const float* __restrict__ z1,
                                             const float* __restrict__ z2,
                                             _Float16* __restrict__ znh) {
    int row = blockIdx.x;
    const float* src = (row < NN) ? (z1 + (size_t)row * DD)
                                  : (z2 + (size_t)(row - NN) * DD);
    int t = threadIdx.x;
    float x = src[t];
    __shared__ float red[128];
    red[t] = x * x;
    __syncthreads();
    for (int s = 64; s > 0; s >>= 1) { if (t < s) red[t] += red[t + s]; __syncthreads(); }
    float inv = 1.0f / fmaxf(sqrtf(red[0]), 1e-12f);
    znh[(size_t)row * DD + t] = (_Float16)(x * inv);
}

__global__ __launch_bounds__(128) void klab(const int* __restrict__ labels,
                                            unsigned char* __restrict__ lab8,
                                            float* __restrict__ srow) {
    int row = blockIdx.x, t = threadIdx.x;
    const int* src = labels + (size_t)row * AA + t * 4;
    unsigned u0 = src[0], u1 = src[1], u2 = src[2], u3 = src[3];
    ((unsigned*)lab8)[(size_t)row * (AA / 4) + t] =
        (u0 & 0xff) | ((u1 & 0xff) << 8) | ((u2 & 0xff) << 16) | ((u3 & 0xff) << 24);
    __shared__ float red[128];
    red[t] = (float)(u0 + u1 + u2 + u3);
    __syncthreads();
    for (int s = 64; s > 0; s >>= 1) { if (t < s) red[t] += red[t + s]; __syncthreads(); }
    if (t == 0) srow[row] = red[0];
}

__global__ __launch_bounds__(128) void kmain(const _Float16* __restrict__ znh,
                                             const unsigned char* __restrict__ lab8,
                                             const float* __restrict__ srow,
                                             float* __restrict__ rowloss) {
    __shared__ unsigned sLab[16 * (AA / 4)];
    __shared__ float sS[4][16], sP[4][16], sNP[4][16];

    int tid = threadIdx.x;
    int wave = tid >> 5;
    int lane = tid & 31;
    int m = lane & 15;
    int h = lane >> 4;
    int rowbase = blockIdx.x * 16;

    const unsigned* labg = (const unsigned*)lab8;
    for (int k = tid; k < 16 * (AA / 4); k += 128)
        sLab[k] = labg[(size_t)rowbase * (AA / 4) + k];
    __syncthreads();

    HU aF[4];
#pragma unroll
    for (int c = 0; c < 4; c++)
#pragma unroll
        for (int v = 0; v < 8; v++)
            aF[c].u[v] = *(const unsigned*)(znh + (size_t)(rowbase + m) * DD + c * 32 + baseF(v, h));

    float sA[8];
#pragma unroll
    for (int r = 0; r < 8; r++) sA[r] = srow[rowbase + r + 8 * h];

    float accS[8], accP[8], accNP[8];
#pragma unroll
    for (int r = 0; r < 8; r++) { accS[r] = 0.f; accP[r] = 0.f; accNP[r] = 0.f; }

    for (int ctm = wave; ctm < NN / 16; ctm += 4) {
        int colrow = ctm * 16 + m;

        v8i icc = {0, 0, 0, 0, 0, 0, 0, 0};
#pragma unroll
        for (int kc = 0; kc < 8; kc++) {
            IU a1, b1;
#pragma unroll
            for (int v = 0; v < 8; v++) {
                int dq = kc * 16 + (baseI(v, h) >> 2);
                a1.u[v] = sLab[m * (AA / 4) + dq];
                b1.u[v] = labg[(size_t)colrow * (AA / 4) + dq];
            }
            icc = __builtin_amdgcn_wmma_i32_16x16x64_iu8(false, a1.v, false, b1.v,
                                                         icc, false, false);
        }

        v8f d1 = {0, 0, 0, 0, 0, 0, 0, 0};
        v8f d2 = {0, 0, 0, 0, 0, 0, 0, 0};
#pragma unroll
        for (int c = 0; c < 4; c++) {
            HU b1, b2;
#pragma unroll
            for (int v = 0; v < 8; v++) {
                int ke = c * 32 + baseF(v, h);
                b1.u[v] = *(const unsigned*)(znh + (size_t)colrow * DD + ke);
                b2.u[v] = *(const unsigned*)(znh + (size_t)(NN + colrow) * DD + ke);
            }
            d1 = __builtin_amdgcn_wmma_f32_16x16x32_f16(false, aF[c].v, false, b1.v,
                                                        (short)0, d1, false, false);
            d2 = __builtin_amdgcn_wmma_f32_16x16x32_f16(false, aF[c].v, false, b2.v,
                                                        (short)0, d2, false, false);
        }

        float scol = srow[colrow];
#pragma unroll
        for (int r = 0; r < 8; r++) {
            int i = rowbase + r + 8 * h;
            float inter = (float)icc[r];
            float uni = sA[r] + scol - inter;
            bool pos = (inter / (uni + 1e-8f)) >= THR;
            float l1 = (d1[r] - 1.0f) * TAUI;
            float l2 = (d2[r] - 1.0f) * TAUI;
            float e1 = __expf(l1), e2 = __expf(l2);
            bool isSelf = (colrow == i);
            if (!isSelf) accS[r] += e1;
            accS[r] += e2;
            bool p1 = pos && !isSelf;
            bool p2 = pos || isSelf;
            if (p1) { accP[r] += l1; accNP[r] += 1.0f; }
            if (p2) { accP[r] += l2; accNP[r] += 1.0f; }
        }
    }

#pragma unroll
    for (int r = 0; r < 8; r++) {
#pragma unroll
        for (int off = 8; off >= 1; off >>= 1) {
            accS[r]  += __shfl_xor(accS[r],  off, 16);
            accP[r]  += __shfl_xor(accP[r],  off, 16);
            accNP[r] += __shfl_xor(accNP[r], off, 16);
        }
    }
    if (m == 0) {
#pragma unroll
        for (int r = 0; r < 8; r++) {
            sS[wave][r + 8 * h]  = accS[r];
            sP[wave][r + 8 * h]  = accP[r];
            sNP[wave][r + 8 * h] = accNP[r];
        }
    }
    __syncthreads();

    if (tid < 16) {
        float S = 0.f, P = 0.f, NP = 0.f;
#pragma unroll
        for (int w = 0; w < 4; w++) { S += sS[w][tid]; P += sP[w][tid]; NP += sNP[w][tid]; }
        float np = fmaxf(NP, 1.0f);
        rowloss[rowbase + tid] = logf(S + 1e-8f) - P / np;
    }
}

__global__ __launch_bounds__(256) void kreduce(const float* __restrict__ rowloss,
                                               float* __restrict__ out) {
    __shared__ float red[256];
    int t = threadIdx.x;
    float s = 0.f;
    for (int i = t; i < NN; i += 256) s += rowloss[i];
    red[t] = s;
    __syncthreads();
    for (int k = 128; k > 0; k >>= 1) { if (t < k) red[t] += red[t + k]; __syncthreads(); }
    if (t == 0) out[0] = red[0] / (float)NN;
}

extern "C" void kernel_launch(void* const* d_in, const int* in_sizes, int n_in,
                              void* d_out, int out_size, void* d_ws, size_t ws_size,
                              hipStream_t stream) {
    const float* z1     = (const float*)d_in[0];
    const float* z2     = (const float*)d_in[1];
    const int*   labels = (const int*)d_in[2];

    char* ws = (char*)d_ws;
    _Float16*      znh     = (_Float16*)ws;
    unsigned char* lab8    = (unsigned char*)(ws + (size_t)2 * NN * DD * 2);
    float*         srow    = (float*)(ws + (size_t)2 * NN * DD * 2 + (size_t)NN * AA);
    float*         rowloss = srow + NN;

    knorm  <<<2 * NN, 128, 0, stream>>>(z1, z2, znh);
    klab   <<<NN,     128, 0, stream>>>(labels, lab8, srow);
    kmain  <<<NN / 16, 128, 0, stream>>>(znh, lab8, srow, rowloss);
    kreduce<<<1,      256, 0, stream>>>(rowloss, (float*)d_out);
}
